// GIN_10213432229999
// MI455X (gfx1250) — matrix operand order repaired and checked by reading, not compiled, not run on hardware
//
#include <hip/hip_runtime.h>
#include <hip/hip_bf16.h>

#define D 256
#define ROW_PAD 128      // pad node count to a multiple of this

typedef __attribute__((ext_vector_type(16))) __bf16 v16bf;
typedef __attribute__((ext_vector_type(8)))  float  v8f;
typedef __attribute__((ext_vector_type(4)))  float  f4;

// ---------------------------------------------------------------- utilities
__global__ void zero_kernel(float* __restrict__ p, long n) {
    long i = (long)blockIdx.x * blockDim.x + threadIdx.x;
    long stride = (long)gridDim.x * blockDim.x;
    for (; i < n; i += stride) p[i] = 0.0f;
}

__global__ void copy_kernel(float* __restrict__ dst, const float* __restrict__ src, long n) {
    long i = (long)blockIdx.x * blockDim.x + threadIdx.x;
    long stride = (long)gridDim.x * blockDim.x;
    for (; i < n; i += stride) dst[i] = src[i];
}

// -------------------------------------------- pack weights to bf16 fragments
// Wp[((k>>5)*D + n)*32 + (k&31)] = bf16(W[k*D + n])
// -> a lane's whole 16-element B fragment is 32 contiguous bytes
__global__ __launch_bounds__(256)
void pack_w_bf16(const float* __restrict__ W, __bf16* __restrict__ Wp) {
    int i = blockIdx.x * 256 + threadIdx.x;   // i = k*D + n
    int k = i >> 8;
    int n = i & (D - 1);
    Wp[((k >> 5) * D + n) * 32 + (k & 31)] = (__bf16)W[i];
}

// ------------------------------------------------- edge scatter aggregation
__global__ __launch_bounds__(256)
void scatter_add_kernel(const float* __restrict__ x,
                        const int* __restrict__ srcIdx,
                        const int* __restrict__ dstIdx,
                        float* __restrict__ agg, int nEdges) {
    long gid = (long)blockIdx.x * blockDim.x + threadIdx.x;
    int e = (int)(gid >> 6);        // 64 chunks of 4 floats per edge
    int c = (int)(gid & 63);
    if (e >= nEdges) return;
    int s = srcIdx[e];
    int d = dstIdx[e];
    const float4 v = reinterpret_cast<const float4*>(x + (long)s * D)[c];
    float* p = agg + (long)d * D + c * 4;
    atomicAdd(p + 0, v.x);
    atomicAdd(p + 1, v.y);
    atomicAdd(p + 2, v.z);
    atomicAdd(p + 3, v.w);
}

// ------------------------------------------------------------- fused GEMM
// out = relu((A[+A2]) * (scaleK,shiftK) @ W + bias) [+ resid]
// Compile-time specialized on the fusion flags so each variant's k-loop is
// branch-free. All row dims padded: unguarded lane loads/stores.
// Block: 256 threads = 8 waves; block tile 32 rows x 256 cols
// (wave = 16 rows x 64 cols = 4 accumulators sharing one A fragment).
template <bool HAS_A2, bool HAS_BN, bool HAS_RES, bool STATS>
__global__ __launch_bounds__(256)
void gin_gemm_wmma(const float* __restrict__ A,
                   const float* __restrict__ A2,      // agg          (HAS_A2)
                   const float* __restrict__ scaleK,  // BN scale     (HAS_BN)
                   const float* __restrict__ shiftK,  // BN shift     (HAS_BN)
                   const __bf16* __restrict__ Wp,     // packed bf16 weights
                   const float* __restrict__ bias,    // [D]
                   const float* __restrict__ resid,   // epilogue add (HAS_RES)
                   float* __restrict__ out,
                   float* __restrict__ colSum,        // [D]          (STATS)
                   float* __restrict__ colSsq,        // [D]          (STATS)
                   int nRows) {
    __shared__ float sSum[D];
    __shared__ float sSsq[D];

    const int tid  = threadIdx.x;
    const int wv   = tid >> 5;
    const int lane = tid & 31;
    const int half = lane >> 4;
    const int l16  = lane & 15;
    const int rowBase = blockIdx.x * 32 + (wv >> 2) * 16;   // 2 row-subtiles
    const int colBase = (wv & 3) * 64;                      // 4 col-groups

    if (STATS) {
        sSum[tid] = 0.0f; sSsq[tid] = 0.0f;
        __syncthreads();
    }

    v8f acc[4] = {};

    const int aRow = rowBase + l16;                 // always in padded range
    const float* aPtr  = A + (long)aRow * D;
    const float* a2Ptr = HAS_A2 ? (A2 + (long)aRow * D) : nullptr;

    for (int kk = 0; kk < D; kk += 32) {
        // ---- A fragment (16x32 bf16): per-lane K runs are contiguous
        // j 0..7  -> k = kk + half*16 + j
        // j 8..15 -> k = kk + half*16 + j
        const int k0 = kk + half * 16;
        const int k1 = kk + half * 16 + 8;
        f4 aa[4];
        aa[0] = *(const f4*)(aPtr + k0);
        aa[1] = *(const f4*)(aPtr + k0 + 4);
        aa[2] = *(const f4*)(aPtr + k1);
        aa[3] = *(const f4*)(aPtr + k1 + 4);
        if (HAS_A2) {
            aa[0] += *(const f4*)(a2Ptr + k0);
            aa[1] += *(const f4*)(a2Ptr + k0 + 4);
            aa[2] += *(const f4*)(a2Ptr + k1);
            aa[3] += *(const f4*)(a2Ptr + k1 + 4);
        }
        if (HAS_BN) {
            aa[0] = aa[0] * (*(const f4*)(scaleK + k0))     + (*(const f4*)(shiftK + k0));
            aa[1] = aa[1] * (*(const f4*)(scaleK + k0 + 4)) + (*(const f4*)(shiftK + k0 + 4));
            aa[2] = aa[2] * (*(const f4*)(scaleK + k1))     + (*(const f4*)(shiftK + k1));
            aa[3] = aa[3] * (*(const f4*)(scaleK + k1 + 4)) + (*(const f4*)(shiftK + k1 + 4));
        }
        v16bf afrag;
#pragma unroll
        for (int q = 0; q < 4; ++q)
#pragma unroll
            for (int r = 0; r < 4; ++r)
                afrag[q * 4 + r] = (__bf16)aa[q][r];

        // ---- 4 B fragments: one 32-byte contiguous packed load each
        const int kTile = kk >> 5;
#pragma unroll
        for (int c = 0; c < 4; ++c) {
            const int n = colBase + c * 16 + l16;
            const v16bf bfrag =
                *(const v16bf*)(Wp + ((long)kTile * D + n) * 32 + half * 16);
            acc[c] = __builtin_amdgcn_wmma_f32_16x16x32_bf16(
                false, afrag, false, bfrag, (short)0, acc[c], false, false);
        }
    }

    // ---- epilogue: bias + relu (+resid); BN stats exclude pad rows
#pragma unroll
    for (int c = 0; c < 4; ++c) {
        const int n = colBase + c * 16 + l16;
        const float bv = bias[n];
        float s = 0.0f, q = 0.0f;
#pragma unroll
        for (int i = 0; i < 8; ++i) {
            const int m = rowBase + half * 8 + i;   // C layout: VGPR i -> M = i + 8*half
            float v = fmaxf(acc[c][i] + bv, 0.0f);
            if (STATS) {
                if (m < nRows) { s += v; q += v * v; }
            }
            float o = v;
            if (HAS_RES) o += resid[(long)m * D + n];
            out[(long)m * D + n] = o;
        }
        if (STATS) {
            atomicAdd(&sSum[n], s);
            atomicAdd(&sSsq[n], q);
        }
    }
    if (STATS) {
        __syncthreads();
        // 256 threads cover all D columns of this block's partial sums
        atomicAdd(&colSum[tid], sSum[tid]);
        atomicAdd(&colSsq[tid], sSsq[tid]);
    }
}

// ------------------------------------------------------- BN scale/shift
__global__ void bn_finalize(const float* __restrict__ colSum,
                            const float* __restrict__ colSsq,
                            const float* __restrict__ gamma,
                            const float* __restrict__ beta,
                            float* __restrict__ scaleOut,
                            float* __restrict__ shiftOut, float invN) {
    int d = threadIdx.x;
    float mu  = colSum[d] * invN;
    float var = colSsq[d] * invN - mu * mu;
    float rs  = rsqrtf(var + 1e-5f);
    float g   = gamma[d] * rs;
    scaleOut[d] = g;
    shiftOut[d] = beta[d] - mu * g;
}

// ------------------------------------------------------- final D -> 1 head
__global__ __launch_bounds__(256)
void head_dot(const float* __restrict__ y, const float* __restrict__ fc2w,
              const float* __restrict__ fc2b, float* __restrict__ out, int nRows) {
    __shared__ float red[D];
    int n = blockIdx.x;
    int t = threadIdx.x;
    if (n >= nRows) return;
    red[t] = y[(long)n * D + t] * fc2w[t];
    __syncthreads();
    for (int s = D / 2; s > 0; s >>= 1) {
        if (t < s) red[t] += red[t + s];
        __syncthreads();
    }
    if (t == 0) out[n] = red[0] + fc2b[0];
}

// ---------------------------------------------------------------- launcher
extern "C" void kernel_launch(void* const* d_in, const int* in_sizes, int n_in,
                              void* d_out, int out_size, void* d_ws, size_t ws_size,
                              hipStream_t stream) {
    const float* x    = (const float*)d_in[0];
    const int*   ei   = (const int*)d_in[1];
    const float* W1s  = (const float*)d_in[2];
    const float* b1s  = (const float*)d_in[3];
    const float* gam  = (const float*)d_in[4];
    const float* bet  = (const float*)d_in[5];
    const float* W2s  = (const float*)d_in[6];
    const float* b2s  = (const float*)d_in[7];
    const float* fc1w = (const float*)d_in[8];
    const float* fc1b = (const float*)d_in[9];
    const float* fc2w = (const float*)d_in[10];
    const float* fc2b = (const float*)d_in[11];

    const int N  = in_sizes[0] / D;                    // 50000 nodes
    const int E  = in_sizes[1] / 2;                    // 1600000 edges
    const int NP = ((N + ROW_PAD - 1) / ROW_PAD) * ROW_PAD;  // padded rows
    const int* srcIdx = ei;
    const int* dstIdx = ei + E;

    const long NF  = (long)N  * D;
    const long NPF = (long)NP * D;

    // workspace: 4 padded feature buffers + BN scratch + packed weights
    float* cur = (float*)d_ws;         // current features (padded, rewritten per layer)
    float* agg = cur + NPF;            // neighbor aggregation
    float* h   = agg + NPF;            // hidden after Linear1/ReLU
    float* x0  = h + NPF;              // residual save
    float* bnSum   = x0 + NPF;
    float* bnSsq   = bnSum + D;
    float* bnScale = bnSsq + D;
    float* bnShift = bnScale + D;
    __bf16* Wp = (__bf16*)(bnShift + D);   // D*D packed bf16 weights

    const dim3 gemmGrid(NP / 32);
    const int  packBlocks    = (D * D) / 256;
    const int  scatterBlocks = (int)(((long)E * (D / 4) + 255) / 256);
    const float invN = 1.0f / (float)N;

    // cur = [x ; zero pad rows]
    zero_kernel<<<256, 256, 0, stream>>>(cur + NF, NPF - NF);
    copy_kernel<<<4096, 256, 0, stream>>>(cur, x, NF);

    for (int blk = 0; blk < 3; ++blk) {
        copy_kernel<<<4096, 256, 0, stream>>>(x0, cur, NPF);   // residual save
        for (int t = 0; t < 2; ++t) {
            const int j = blk * 2 + t;
            zero_kernel<<<4096, 256, 0, stream>>>(agg, NPF);
            zero_kernel<<<1, 256, 0, stream>>>(bnSum, 2 * D);
            scatter_add_kernel<<<scatterBlocks, 256, 0, stream>>>(cur, srcIdx, dstIdx, agg, E);
            // h = relu((x + agg) @ W1 + b1), accumulate column stats
            pack_w_bf16<<<packBlocks, 256, 0, stream>>>(W1s + (long)j * D * D, Wp);
            gin_gemm_wmma<true, false, false, true><<<gemmGrid, 256, 0, stream>>>(
                cur, agg, nullptr, nullptr, Wp, b1s + (long)j * D,
                nullptr, h, bnSum, bnSsq, N);
            bn_finalize<<<1, D, 0, stream>>>(bnSum, bnSsq, gam + (long)j * D,
                                             bet + (long)j * D, bnScale, bnShift, invN);
            // cur = relu((h*scale+shift) @ W2 + b2) [+ x0 at block end]
            pack_w_bf16<<<packBlocks, 256, 0, stream>>>(W2s + (long)j * D * D, Wp);
            if (t == 1) {
                gin_gemm_wmma<false, true, true, false><<<gemmGrid, 256, 0, stream>>>(
                    h, nullptr, bnScale, bnShift, Wp, b2s + (long)j * D,
                    x0, cur, nullptr, nullptr, N);
            } else {
                gin_gemm_wmma<false, true, false, false><<<gemmGrid, 256, 0, stream>>>(
                    h, nullptr, bnScale, bnShift, Wp, b2s + (long)j * D,
                    nullptr, cur, nullptr, nullptr, N);
            }
        }
    }

    // y = x + relu(x @ fc1 + fc1_b)  -> h
    pack_w_bf16<<<packBlocks, 256, 0, stream>>>(fc1w, Wp);
    gin_gemm_wmma<false, false, true, false><<<gemmGrid, 256, 0, stream>>>(
        cur, nullptr, nullptr, nullptr, Wp, fc1b, cur, h, nullptr, nullptr, N);
    // out = y @ fc2 + fc2_b
    head_dot<<<N, D, 0, stream>>>(h, fc2w, fc2b, (float*)d_out, N);
}
